// ConvLayer_42142219109069
// MI455X (gfx1250) — matrix operand order repaired and checked by reading, not compiled, not run on hardware
//
#include <hip/hip_runtime.h>
#include <cstdint>
#include <cstddef>

#define DIM 128  // feature dim D

// ---------- CDNA5 WMMA types ----------
typedef __attribute__((ext_vector_type(16))) __bf16 v16bf;
typedef __attribute__((ext_vector_type(8)))  float  v8f;
typedef __attribute__((ext_vector_type(4)))  unsigned int v4u;

union ABFrag { v16bf v; v4u q[2]; };

__device__ __forceinline__ __bf16 f2bf(float x) {
  // round-to-nearest-even f32 -> bf16, storage-only use of __bf16
  unsigned u = __builtin_bit_cast(unsigned, x);
  u += 0x7FFFu + ((u >> 16) & 1u);
  unsigned short h = (unsigned short)(u >> 16);
  return __builtin_bit_cast(__bf16, h);
}

// =====================================================================
// GEMM: C[N,M] (f32) = A[N,K] (bf16 row-major) * Bt[M,K]^T (bf16 row-major,
// i.e. Bt[m][k] = W[k][m]).
// Block = 8 waves; block stages a (MT*32 cols x K) B panel in LDS once
// (dynamic LDS, padded rows -> conflict-free ds_load_b128), then each wave
// computes a 16 x (MT*32) tile: per K-step one global A fragment feeds
// 2*MT v_wmma_f32_16x16x32_bf16 ops with B fragments read from LDS.
// Grid: x = ceil((N/16)/8), y = M/(MT*32).
// Requires N % 16 == 0, K % 32 == 0, M % (MT*32) == 0.
// =====================================================================
template <int MT>
__global__ __launch_bounds__(256) void k_gemm_bf16(
    const __bf16* __restrict__ A, const __bf16* __restrict__ Bt,
    float* __restrict__ C, int N, int K, int M)
{
  extern __shared__ __bf16 ldsB[];        // (MT*32) rows x (K+8) bf16
  const int PAD = 8;                      // 16B row pad -> 4-bank row skew
  const int ldsStride = K + PAD;
  const int c0 = blockIdx.y * (MT * 32);

  // ---- cooperative global -> LDS copy of the B panel (16B chunks) ----
  {
    const int chunksPerRow = K >> 3;                  // 8 bf16 per 16B
    const int totalChunks  = (MT * 32) * chunksPerRow;
    for (int idx = threadIdx.x; idx < totalChunks; idx += 256) {
      int row = idx / chunksPerRow;
      int ck  = idx - row * chunksPerRow;
      v4u d = *(const v4u*)(Bt + (size_t)(c0 + row) * K + ck * 8);
      *(v4u*)(ldsB + (size_t)row * ldsStride + ck * 8) = d;
    }
  }
  __syncthreads();

  const int lane = threadIdx.x & 31;
  const int wave = threadIdx.x >> 5;
  const int rowTile = blockIdx.x * 8 + wave;
  if (rowTile * 16 >= N) return;          // wave-uniform; no barriers follow
  const int r0 = rowTile << 4;
  const int half = lane >> 4;             // lane group 0..15 / 16..31
  const int l16  = lane & 15;

  // A fragment: lane holds row (r0+l16); elems 0..7 = K[k0+half*8 ..],
  // elems 8..15 = K[k0+16+half*8 ..]  (ISA 16-bit A 16x32 layout)
  const __bf16* aP = A + (size_t)(r0 + l16) * K + half * 8;
  // B fragment (16-col tile t) from LDS: lane holds col (t*16+l16);
  // elems 0..7 -> K = k0 + half*8 + i, elems 8..15 -> K = k0 + 16 + half*8 + (i-8)
  const __bf16* bL = ldsB + (size_t)l16 * ldsStride + half * 8;

  v8f acc[2 * MT];
#pragma unroll
  for (int t = 0; t < 2 * MT; ++t) acc[t] = (v8f){};

  for (int k0 = 0; k0 < K; k0 += 32) {
    ABFrag a;
    a.q[0] = *(const v4u*)(aP + k0);
    a.q[1] = *(const v4u*)(aP + k0 + 16);
    ABFrag b[2 * MT];
#pragma unroll
    for (int t = 0; t < 2 * MT; ++t) {
      const __bf16* bp = bL + (size_t)(t * 16) * ldsStride + k0;
      b[t].q[0] = *(const v4u*)(bp);
      b[t].q[1] = *(const v4u*)(bp + 16);
    }
#pragma unroll
    for (int t = 0; t < 2 * MT; ++t)
      acc[t] = __builtin_amdgcn_wmma_f32_16x16x32_bf16(false, a.v, false, b[t].v,
                                                       (short)0, acc[t], false, false);
  }
  // C/D layout: elem i of lane L -> row r0 + (L>>4)*8 + i, col tile + (L&15)
  float* cP = C + (size_t)(r0 + half * 8) * M + c0 + l16;
#pragma unroll
  for (int t = 0; t < 2 * MT; ++t) {
#pragma unroll
    for (int i = 0; i < 8; ++i)
      cP[(size_t)i * M + t * 16] = acc[t][i];
  }
}

// ---------- weight convert + transpose: W[K,M] f32 -> Wt[M,K] bf16 ----------
__global__ void k_cvt_w_t(const float* __restrict__ W, __bf16* __restrict__ Wt,
                          int K, int M)
{
  long i = (long)blockIdx.x * blockDim.x + threadIdx.x;
  if (i >= (long)K * M) return;
  int k = (int)(i / M), m = (int)(i % M);
  Wt[(size_t)m * K + k] = f2bf(W[i]);
}

// ---------- zero fill ----------
__global__ void k_zero_f32(float* __restrict__ p, long n)
{
  long i = (long)blockIdx.x * blockDim.x + threadIdx.x;
  if (i < n) p[i] = 0.0f;
}

// ---------- per-column sum / sumsq partial reduction ----------
__global__ void k_colstats(const float* __restrict__ C,
                           float* __restrict__ sum, float* __restrict__ sq,
                           long N, int M, int rowsPerBlock)
{
  const int col = threadIdx.x % M;
  const int g   = threadIdx.x / M;
  const int G   = (int)blockDim.x / M;
  long r0 = (long)blockIdx.x * rowsPerBlock;
  long r1 = r0 + rowsPerBlock; if (r1 > N) r1 = N;
  float s = 0.0f, q = 0.0f;
  for (long r = r0 + g; r < r1; r += G) {
    float v = C[r * M + col];
    s += v; q += v * v;
  }
  atomicAdd(&sum[col], s);
  atomicAdd(&sq[col],  q);
}

// ---------- finalize BN: scale = g*rsqrt(var+eps), shift = b - mean*scale ----
__global__ void k_bnfinal(const float* __restrict__ sum, const float* __restrict__ sq,
                          const float* __restrict__ g, const float* __restrict__ b,
                          float* __restrict__ scale, float* __restrict__ shift,
                          int M, float invN)
{
  int m = blockIdx.x * blockDim.x + threadIdx.x;
  if (m >= M) return;
  float mean = sum[m] * invN;
  float var  = sq[m] * invN - mean * mean;
  float sc   = g[m] * rsqrtf(var + 1e-5f);
  scale[m] = sc;
  shift[m] = b[m] - mean * sc;
}

// ---------- BN + ReLU, optional bf16 and f32 outputs (strided) ----------
__global__ void k_bnrelu(const float* __restrict__ C,
                         const float* __restrict__ scale, const float* __restrict__ shift,
                         long N, int M,
                         __bf16* __restrict__ obf, int obfStride,
                         float* __restrict__ of32, int of32Stride)
{
  long i = (long)blockIdx.x * blockDim.x + threadIdx.x;
  if (i >= N * M) return;
  long r = i / M; int c = (int)(i % M);
  float v = C[i] * scale[c] + shift[c];
  v = v > 0.0f ? v : 0.0f;
  if (obf)  obf[r * (long)obfStride + c]   = f2bf(v);
  if (of32) of32[r * (long)of32Stride + c] = v;
}

// ---------- X0 = [edge_rep | node_rep[u]+node_rep[v]] (bf16, E x 256) -------
__global__ void k_concat_edge0(const float* __restrict__ node, const float* __restrict__ edge,
                               const int* __restrict__ eidx, __bf16* __restrict__ X, long E)
{
  long i = (long)blockIdx.x * blockDim.x + threadIdx.x;
  if (i >= E * DIM) return;
  long e = i >> 7; int c = (int)(i & (DIM - 1));
  int uu = eidx[2 * e], vv = eidx[2 * e + 1];
  X[e * 256 + c]       = f2bf(edge[i]);
  X[e * 256 + 128 + c] = f2bf(node[(size_t)uu * DIM + c] + node[(size_t)vv * DIM + c]);
}

// ---------- scatter-add src[e] into dst[u] and dst[v] -----------------------
__global__ void k_scatter_edge2(const float* __restrict__ src, const int* __restrict__ eidx,
                                float* __restrict__ dst, long E)
{
  long i = (long)blockIdx.x * blockDim.x + threadIdx.x;
  if (i >= E * DIM) return;
  long e = i >> 7; int c = (int)(i & (DIM - 1));
  float v = src[i];
  atomicAdd(&dst[(size_t)eidx[2 * e]     * DIM + c], v);
  atomicAdd(&dst[(size_t)eidx[2 * e + 1] * DIM + c], v);
}

// ---------- X = [a | b] (bf16, N x 256) --------------------------------------
__global__ void k_concat2(const float* __restrict__ a, const float* __restrict__ b,
                          __bf16* __restrict__ X, long N)
{
  long i = (long)blockIdx.x * blockDim.x + threadIdx.x;
  if (i >= N * DIM) return;
  long r = i >> 7; int c = (int)(i & (DIM - 1));
  X[r * 256 + c]       = f2bf(a[i]);
  X[r * 256 + 128 + c] = f2bf(b[i]);
}

// ---------- e2c gather + first 2 thirds of Xc --------------------------------
__global__ void k_e2c(const float* __restrict__ ean, const int* __restrict__ atomNode,
                      const float* __restrict__ cyc, float* __restrict__ e2c,
                      __bf16* __restrict__ Xc, long A)
{
  long i = (long)blockIdx.x * blockDim.x + threadIdx.x;
  if (i >= A * DIM) return;
  long a = i >> 7; int c = (int)(i & (DIM - 1));
  float x = ean[(size_t)atomNode[a] * DIM + c];
  e2c[i] = x;
  Xc[a * 384 + c]       = f2bf(cyc[i]);
  Xc[a * 384 + 128 + c] = f2bf(x);
}

// ---------- generic scatter-add by index array -------------------------------
__global__ void k_scatter_idx(const float* __restrict__ src, const int* __restrict__ idx,
                              float* __restrict__ dst, long n)
{
  long i = (long)blockIdx.x * blockDim.x + threadIdx.x;
  if (i >= n * DIM) return;
  long a = i >> 7; int c = (int)(i & (DIM - 1));
  atomicAdd(&dst[(size_t)idx[a] * DIM + c], src[i]);
}

// ---------- last third of Xc: cyc_sum[cycle_id] ------------------------------
__global__ void k_gather384(const float* __restrict__ cycsum, const int* __restrict__ cid,
                            __bf16* __restrict__ Xc, long A)
{
  long i = (long)blockIdx.x * blockDim.x + threadIdx.x;
  if (i >= A * DIM) return;
  long a = i >> 7; int c = (int)(i & (DIM - 1));
  Xc[a * 384 + 256 + c] = f2bf(cycsum[(size_t)cid[a] * DIM + c]);
}

// ---------- Xe2 = [edge_rep | node_acc[u]+node_acc[v]] -----------------------
__global__ void k_concat_gather2(const float* __restrict__ edge, const float* __restrict__ acc,
                                 const int* __restrict__ eidx, __bf16* __restrict__ X, long E)
{
  long i = (long)blockIdx.x * blockDim.x + threadIdx.x;
  if (i >= E * DIM) return;
  long e = i >> 7; int c = (int)(i & (DIM - 1));
  int uu = eidx[2 * e], vv = eidx[2 * e + 1];
  X[e * 256 + c]       = f2bf(edge[i]);
  X[e * 256 + 128 + c] = f2bf(acc[(size_t)uu * DIM + c] + acc[(size_t)vv * DIM + c]);
}

// =====================================================================
// host-side helpers
// =====================================================================
static inline unsigned nblk(long n, int b) { return (unsigned)((n + b - 1) / b); }

static void run_mlp(hipStream_t s, const __bf16* X, long N, int Kin,
                    const __bf16* W1t, const float* g1, const float* b1,
                    const __bf16* W2t, const float* g2, const float* b2,
                    float* G, __bf16* Hb,
                    float* ssum, float* ssq, float* sscale, float* sshift,
                    __bf16* obf, int obfStride, float* of32, int of32Stride)
{
  const int H = 256, DO = 128;
  const int CH = 512;                 // rows per colstats block
  // layer 1: G = X @ W1  (N x H), wave tile 16x128 -> grid.y = 2
  {
    dim3 g((unsigned)((N / 16 + 7) / 8), H / 128);
    size_t lds = (size_t)128 * (Kin + 8) * 2;
    k_gemm_bf16<4><<<g, dim3(256), lds, s>>>(X, W1t, G, (int)N, Kin, H);
  }
  k_zero_f32<<<nblk(512, 256), 256, 0, s>>>(ssum, 512);   // sum+sq contiguous
  k_colstats<<<nblk(N, CH), 256, 0, s>>>(G, ssum, ssq, N, H, CH);
  k_bnfinal<<<1, 256, 0, s>>>(ssum, ssq, g1, b1, sscale, sshift, H, 1.0f / (float)N);
  k_bnrelu<<<nblk(N * H, 256), 256, 0, s>>>(G, sscale, sshift, N, H, Hb, H,
                                            (float*)nullptr, 0);
  // layer 2: G = Hb @ W2  (N x 128), wave tile 16x128 -> grid.y = 1
  {
    dim3 g((unsigned)((N / 16 + 7) / 8), DO / 128);
    size_t lds = (size_t)128 * (H + 8) * 2;
    k_gemm_bf16<4><<<g, dim3(256), lds, s>>>(Hb, W2t, G, (int)N, H, DO);
  }
  k_zero_f32<<<nblk(512, 256), 256, 0, s>>>(ssum, 512);
  k_colstats<<<nblk(N, CH), 256, 0, s>>>(G, ssum, ssq, N, DO, CH);
  k_bnfinal<<<1, 128, 0, s>>>(ssum, ssq, g2, b2, sscale, sshift, DO, 1.0f / (float)N);
  k_bnrelu<<<nblk(N * DO, 256), 256, 0, s>>>(G, sscale, sshift, N, DO,
                                             obf, obfStride, of32, of32Stride);
}

extern "C" void kernel_launch(void* const* d_in, const int* in_sizes, int n_in,
                              void* d_out, int out_size, void* d_ws, size_t ws_size,
                              hipStream_t stream)
{
  (void)n_in; (void)out_size; (void)ws_size;
  const float* node_rep   = (const float*)d_in[0];
  const float* edge_rep   = (const float*)d_in[1];
  const float* cycle_rep  = (const float*)d_in[2];
  const int*   edge_index = (const int*)d_in[3];
  const int*   atom_node  = (const int*)d_in[4];
  const int*   cycle_id   = (const int*)d_in[5];

  const long nN = in_sizes[0] / DIM;   // 100000 nodes
  const long nE = in_sizes[1] / DIM;   // 200000 edges
  const long nA = in_sizes[2] / DIM;   // 220000 cycle atoms
  const long nC = 40000;               // NUM_CYCLES

  // params: 5 MLPs x {W1,g1,b1,W2,g2,b2} starting at d_in[6]
  const int K1[5] = {256, 256, 384, 256, 256};
  const float *W1[5], *g1[5], *b1[5], *W2[5], *g2[5], *b2[5];
  for (int m = 0; m < 5; ++m) {
    int base = 6 + m * 6;
    W1[m] = (const float*)d_in[base + 0];
    g1[m] = (const float*)d_in[base + 1];
    b1[m] = (const float*)d_in[base + 2];
    W2[m] = (const float*)d_in[base + 3];
    g2[m] = (const float*)d_in[base + 4];
    b2[m] = (const float*)d_in[base + 5];
  }

  // ---------------- workspace carve (phase-overlapped) ----------------
  size_t cur = 0;
  auto alloc = [&](size_t bytes) -> void* {
    void* p = (char*)d_ws + cur;
    cur = (cur + bytes + 255) & ~(size_t)255;
    return p;
  };
  __bf16* W1t[5]; __bf16* W2t[5];
  for (int m = 0; m < 5; ++m) {
    W1t[m] = (__bf16*)alloc((size_t)K1[m] * 256 * 2);
    W2t[m] = (__bf16*)alloc((size_t)128 * 256 * 2);
  }
  float* stat   = (float*)alloc(4 * 256 * sizeof(float));
  float* ssum   = stat;        float* ssq    = stat + 256;
  float* sscale = stat + 512;  float* sshift = stat + 768;

  float*  G  = (float*)alloc((size_t)nA * 256 * 4);    // GEMM out scratch
  __bf16* Hb = (__bf16*)alloc((size_t)nA * 256 * 2);   // hidden activations
  __bf16* Xf = (__bf16*)alloc((size_t)nE * 256 * 2);   // [edge_out1|edge_out2] bf16
  char*   R4 = (char*)alloc((size_t)nE * 256 * 2);     // reused region
  char*   R5 = (char*)alloc((size_t)nA * DIM * 4);     // reused region
  char*   R6 = (char*)alloc((size_t)nA * 384 * 2);     // reused region

  float* node_out  = (float*)d_out;
  float* edge_out  = (float*)d_out + (size_t)nN * DIM;
  float* cycle_out = (float*)d_out + ((size_t)nN + (size_t)nE) * DIM;

  // ---------------- weight convert + transpose ----------------
  for (int m = 0; m < 5; ++m) {
    long n1 = (long)K1[m] * 256;
    k_cvt_w_t<<<nblk(n1, 256), 256, 0, stream>>>(W1[m], W1t[m], K1[m], 256);
    k_cvt_w_t<<<nblk(256 * 128, 256), 256, 0, stream>>>(W2[m], W2t[m], 256, 128);
  }

  // ---------------- phase A: edge_mlp_0 ----------------
  __bf16* X0 = (__bf16*)R4;
  k_concat_edge0<<<nblk(nE * DIM, 256), 256, 0, stream>>>(node_rep, edge_rep,
                                                          edge_index, X0, nE);
  float* edge_out1 = (float*)R5;
  run_mlp(stream, X0, nE, 256, W1t[0], g1[0], b1[0], W2t[0], g2[0], b2[0],
          G, Hb, ssum, ssq, sscale, sshift,
          /*bf16 -> Xf cols 0:128*/ Xf, 256, /*f32*/ edge_out1, DIM);

  // ---------------- phase B: node_mlp ----------------
  float* edge2node = (float*)R4;
  k_zero_f32<<<nblk(nN * DIM, 256), 256, 0, stream>>>(edge2node, nN * DIM);
  k_scatter_edge2<<<nblk(nE * DIM, 256), 256, 0, stream>>>(edge_out1, edge_index,
                                                           edge2node, nE);
  __bf16* Xn = (__bf16*)(R4 + (size_t)nN * DIM * 4);
  k_concat2<<<nblk(nN * DIM, 256), 256, 0, stream>>>(node_rep, edge2node, Xn, nN);
  run_mlp(stream, Xn, nN, 256, W1t[1], g1[1], b1[1], W2t[1], g2[1], b2[1],
          G, Hb, ssum, ssq, sscale, sshift,
          (__bf16*)nullptr, 0, node_out, DIM);

  // ---------------- phase C: cycle_mlp ----------------
  float* ean = (float*)R4;                                 // edge_at_node
  k_zero_f32<<<nblk(nN * DIM, 256), 256, 0, stream>>>(ean, nN * DIM);
  k_scatter_edge2<<<nblk(nE * DIM, 256), 256, 0, stream>>>(edge_rep, edge_index,
                                                           ean, nE);
  float*  e2c = (float*)R5;
  __bf16* Xc  = (__bf16*)R6;
  k_e2c<<<nblk(nA * DIM, 256), 256, 0, stream>>>(ean, atom_node, cycle_rep,
                                                 e2c, Xc, nA);
  float* cyc_sum = (float*)(R4 + (size_t)nN * DIM * 4);
  k_zero_f32<<<nblk(nC * DIM, 256), 256, 0, stream>>>(cyc_sum, nC * DIM);
  k_scatter_idx<<<nblk(nA * DIM, 256), 256, 0, stream>>>(e2c, cycle_id, cyc_sum, nA);
  k_gather384<<<nblk(nA * DIM, 256), 256, 0, stream>>>(cyc_sum, cycle_id, Xc, nA);
  run_mlp(stream, Xc, nA, 384, W1t[2], g1[2], b1[2], W2t[2], g2[2], b2[2],
          G, Hb, ssum, ssq, sscale, sshift,
          (__bf16*)nullptr, 0, cycle_out, DIM);

  // ---------------- phase D: edge_mlp_ec ----------------
  float* node_acc = (float*)R4;
  k_zero_f32<<<nblk(nN * DIM, 256), 256, 0, stream>>>(node_acc, nN * DIM);
  k_scatter_idx<<<nblk(nA * DIM, 256), 256, 0, stream>>>(cycle_out, atom_node,
                                                         node_acc, nA);
  __bf16* Xe2 = (__bf16*)R6;
  k_concat_gather2<<<nblk(nE * DIM, 256), 256, 0, stream>>>(edge_rep, node_acc,
                                                            edge_index, Xe2, nE);
  run_mlp(stream, Xe2, nE, 256, W1t[3], g1[3], b1[3], W2t[3], g2[3], b2[3],
          G, Hb, ssum, ssq, sscale, sshift,
          /*bf16 -> Xf cols 128:256*/ Xf + 128, 256, (float*)nullptr, 0);

  // ---------------- phase E: edge_mlp_final ----------------
  run_mlp(stream, Xf, nE, 256, W1t[4], g1[4], b1[4], W2t[4], g2[4], b2[4],
          G, Hb, ssum, ssq, sscale, sshift,
          (__bf16*)nullptr, 0, edge_out, DIM);
}
